// RwkvEncoder_83588653515153
// MI455X (gfx1250) — matrix operand order repaired and checked by reading, not compiled, not run on hardware
//
#include <hip/hip_runtime.h>
#include <hip/hip_bf16.h>
#include <cstddef>
#include <cstdint>
#include <type_traits>

// ---------------- model constants ----------------
constexpr int Vn = 65536, Cn = 512, Ln = 12, Fn = 2048, HSn = 64;
constexpr int Hn = Cn / HSn;           // 8
constexpr int TMIX = 32, TDEC = 64;
constexpr int Bn = 4, Tn = 512;
constexpr int BT = Bn * Tn;            // 2048
constexpr float GN_EPS = 1e-5f * 64.f;

typedef __attribute__((ext_vector_type(16))) _Float16 v16h;
typedef __attribute__((ext_vector_type(8)))  _Float16 v8h;
typedef __attribute__((ext_vector_type(8)))  float    v8f;
typedef __attribute__((ext_vector_type(4)))  unsigned int v4u;
typedef __attribute__((ext_vector_type(4)))  int      v4i;
typedef __attribute__((ext_vector_type(8)))  int      v8i;

enum { ACT_NONE = 0, ACT_TANH = 1, ACT_SIGMOID = 2, ACT_SILU = 3, ACT_RELUSQ = 4 };

// Issue one TDM tensor_load_to_lds for a 2D f16 tile [rows x 32] with row
// stride ldElems (elements), destination LDS rows padded 64B data + 16B pad.
__device__ __forceinline__ void tdm_load_tile(unsigned ldsAddr, uint64_t gaddr,
                                              int rows, int ldElems)
{
    v4u g0 = { 1u,                                  // count=1 (valid descriptor)
               ldsAddr,
               (unsigned)gaddr,
               (unsigned)(gaddr >> 32) | (2u << 30) };   // addr[56:32] | type=2
    // group1: data_size=1(2B), pad_enable, pad_interval=3(64B), pad_amount=3(16B)
    unsigned w0 = (1u << 16) | (1u << 20) | (3u << 22) | (3u << 25);
    unsigned w1 = (32u << 16);                 // tensor_dim0 = 32 (tile-local)
    unsigned w2 = ((unsigned)rows << 16);      // tensor_dim1 = rows
    unsigned w3 = (32u << 16);                 // tile_dim0 = 32
    unsigned w4 = (unsigned)rows;              // tile_dim1 = rows (tile_dim2=0)
    unsigned w5 = (unsigned)ldElems;           // tensor_dim0_stride[31:0]
    v8i g1 = { (int)w0, (int)w1, (int)w2, (int)w3, (int)w4, (int)w5, 0, 0 };
    v4i z4 = { 0, 0, 0, 0 };
#if defined(__clang_major__) && (__clang_major__ >= 23)
    v8i z8 = { 0, 0, 0, 0, 0, 0, 0, 0 };
    __builtin_amdgcn_tensor_load_to_lds(g0, g1, z4, z4, z8, 0);
#else
    __builtin_amdgcn_tensor_load_to_lds(g0, g1, z4, z4, 0);
#endif
}

// ---------------- pipelined WMMA GEMM (async + TDM staged) ----------------
// C[M x N] (+)= act(A[M x K] @ B[K x N])
// A: f16 row-major [M][lda];  Bt: f16 TRANSPOSED [N][ldb]
// Block: 256 threads (8 wave32). Block tile: (64*TMW) x (32*TNW).
// Wave (wm,wn) computes (16*TMW) x (16*TNW) via TMW*TNW WMMAs per K-step.
// A tiles: global_load_async_to_lds_b128 (ASYNCcnt). B tiles: TDM (TENSORcnt).
// 3-stage LDS pipeline, 1 stage issued ahead, two barriers per K-step.
template <int ACT, bool ACCUM, bool HOUT, int TMW, int TNW>
__global__ __launch_bounds__(256)
void wmma_gemm(const _Float16* __restrict__ A, int lda,
               const _Float16* __restrict__ Bt, int ldb,
               void* __restrict__ Cv, int ldc, int K)
{
    using OutT = typename std::conditional<HOUT, _Float16, float>::type;
    OutT* Cc = (OutT*)Cv;
    constexpr int BM = 64 * TMW;
    constexpr int BN = 32 * TNW;

    __shared__ alignas(16) _Float16 As[3][BM][40];   // 64B data + 16B pad rows
    __shared__ alignas(16) _Float16 Bs[3][BN][40];   // [n][k]

    const int tid  = threadIdx.x;
    const int lane = tid & 31;
    const int wave = tid >> 5;
    const int wm   = wave >> 1;              // 0..3
    const int wn   = wave & 1;               // 0..1
    const int m0   = blockIdx.y * BM;
    const int n0   = blockIdx.x * BN;
    const int lrow = lane & 15;
    const int lhi  = lane >> 4;              // 0 or 1

    const unsigned asOff = (unsigned)(uint64_t)(uintptr_t)&As[0][0][0];
    const unsigned bsOff = (unsigned)(uint64_t)(uintptr_t)&Bs[0][0][0];

    v8f acc[TMW][TNW];
    #pragma unroll
    for (int i = 0; i < TMW; ++i)
        #pragma unroll
        for (int j = 0; j < TNW; ++j) acc[i][j] = (v8f){};

    const int nk = K / 32;

    // stage issue: A via async b128 (TMW instr/wave), B via one TDM (wave 0)
    auto issue = [&](int st, int ki) {
        int kb = ki * 32;
        #pragma unroll
        for (int q = 0; q < TMW; ++q) {
            int chunk = tid + q * 256;               // 8-half chunks of A tile
            int r  = chunk >> 2;
            int cg = (chunk & 3) * 8;
            unsigned aLds = asOff + (unsigned)((st * BM + r) * 40 + cg) * 2;
            uint64_t ga = (uint64_t)(uintptr_t)(A + (size_t)(m0 + r) * lda + kb + cg);
            asm volatile("global_load_async_to_lds_b128 %0, %1, off"
                         :: "v"(aLds), "v"(ga) : "memory");
        }
        if (wave == 0) {
            unsigned bLds = bsOff + (unsigned)(st * BN * 40) * 2;
            uint64_t gb = (uint64_t)(uintptr_t)(Bt + (size_t)n0 * ldb + kb);
            tdm_load_tile(bLds, gb, BN, ldb);
        }
        if (ki + 1 < nk)   // warm L2 for next A tile
            __builtin_prefetch(A + (size_t)(m0 + (tid >> 2)) * lda + kb + 32, 0, 1);
    };

    issue(0, 0);
    for (int i = 0; i < nk; ++i) {
        const int st = i % 3;
        if (i + 1 < nk) {
            issue((i + 1) % 3, i + 1);
            if (TMW == 2) asm volatile("s_wait_asynccnt 0x2" ::: "memory");
            else          asm volatile("s_wait_asynccnt 0x1" ::: "memory");
            if (wave == 0) __builtin_amdgcn_s_wait_tensorcnt(1);
        } else {
            asm volatile("s_wait_asynccnt 0x0" ::: "memory");
            if (wave == 0) __builtin_amdgcn_s_wait_tensorcnt(0);
        }
        __syncthreads();

        // A fragments: lanes 0-15 K{0..7,16..23}, lanes 16-31 K{8..15,24..31}
        v16h afr[TMW];
        #pragma unroll
        for (int sm = 0; sm < TMW; ++sm) {
            const int arow = wm * 16 * TMW + sm * 16 + lrow;
            v8h a_lo = *(const v8h*)&As[st][arow][lhi * 8];
            v8h a_hi = *(const v8h*)&As[st][arow][16 + lhi * 8];
            afr[sm] = __builtin_shufflevector(a_lo, a_hi,
                        0,1,2,3,4,5,6,7,8,9,10,11,12,13,14,15);
        }
        // B fragments: lane = column, lanes 0-15 K{0..7,16..23}, lanes 16-31 K{8..15,24..31}
        v16h bfr[TNW];
        #pragma unroll
        for (int sn = 0; sn < TNW; ++sn) {
            const int col = wn * 16 * TNW + sn * 16 + lrow;
            v8h b_lo = *(const v8h*)&Bs[st][col][lhi * 8];
            v8h b_hi = *(const v8h*)&Bs[st][col][16 + lhi * 8];
            bfr[sn] = __builtin_shufflevector(b_lo, b_hi,
                        0,1,2,3,4,5,6,7,8,9,10,11,12,13,14,15);
        }
        #pragma unroll
        for (int sm = 0; sm < TMW; ++sm)
            #pragma unroll
            for (int sn = 0; sn < TNW; ++sn)
                acc[sm][sn] = __builtin_amdgcn_wmma_f32_16x16x32_f16(
                    false, afr[sm], false, bfr[sn], (short)0, acc[sm][sn],
                    false, false);
        __syncthreads();
    }

    // D layout: VGPR v -> row v + lhi*8; lane&15 -> col
    #pragma unroll
    for (int sm = 0; sm < TMW; ++sm) {
        #pragma unroll
        for (int sn = 0; sn < TNW; ++sn) {
            const int col = n0 + wn * 16 * TNW + sn * 16 + lrow;
            #pragma unroll
            for (int v = 0; v < 8; ++v) {
                int rowg = m0 + wm * 16 * TMW + sm * 16 + v + lhi * 8;
                float x = acc[sm][sn][v];
                if (ACT == ACT_TANH)         x = tanhf(x);
                else if (ACT == ACT_SIGMOID) x = 1.f / (1.f + __expf(-x));
                else if (ACT == ACT_SILU)    x = x / (1.f + __expf(-x));
                else if (ACT == ACT_RELUSQ)  { float t = x > 0.f ? x : 0.f; x = t * t; }
                OutT* dst = Cc + (size_t)rowg * ldc + col;
                if (ACCUM) *dst = (OutT)(*(float*)dst + x);
                else       *dst = (OutT)x;
            }
        }
    }
}

// fp32 [K x N] -> f16 transposed [N x K]; K,N multiples of 32; 256 threads
__global__ void transpose_f16_kernel(const float* __restrict__ in,
                                     _Float16* __restrict__ out, int K, int N)
{
    __shared__ float tile[32][33];
    int nb = blockIdx.x * 32, kb = blockIdx.y * 32;
    int tx = threadIdx.x & 31, ty = threadIdx.x >> 5;   // ty 0..7
    #pragma unroll
    for (int i = ty; i < 32; i += 8)
        tile[i][tx] = in[(size_t)(kb + i) * N + nb + tx];
    __syncthreads();
    #pragma unroll
    for (int i = ty; i < 32; i += 8)
        out[(size_t)(nb + i) * K + kb + tx] = (_Float16)tile[tx][i];
}

// ---------------- elementwise / norm kernels ----------------

__global__ void lens_kernel(const int* __restrict__ idx, int* __restrict__ lens)
{
    __shared__ int red[Tn];
    int b = blockIdx.x, t = threadIdx.x;
    int tok = idx[b * Tn + t];
    red[t] = (tok != 0 && tok != 1) ? 1 : 0;
    __syncthreads();
    for (int s = Tn / 2; s > 0; s >>= 1) {
        if (t < s) red[t] += red[t + s];
        __syncthreads();
    }
    if (t == 0) lens[b] = red[0];
}

__global__ void embed_kernel(const int* __restrict__ idx, const float* __restrict__ emb,
                             float* __restrict__ x)
{
    int row = blockIdx.x, c = threadIdx.x;
    x[(size_t)row * Cn + c] = emb[(size_t)idx[row] * Cn + c];
}

__global__ void layernorm_kernel(const float* __restrict__ in, float* __restrict__ out,
                                 const float* __restrict__ w, const float* __restrict__ b)
{
    __shared__ float red[Cn];
    int row = blockIdx.x, c = threadIdx.x;
    float v = in[(size_t)row * Cn + c];
    red[c] = v; __syncthreads();
    for (int s = Cn / 2; s > 0; s >>= 1) { if (c < s) red[c] += red[c + s]; __syncthreads(); }
    float mu = red[0] * (1.f / Cn);
    __syncthreads();
    float xc = v - mu;
    red[c] = xc * xc; __syncthreads();
    for (int s = Cn / 2; s > 0; s >>= 1) { if (c < s) red[c] += red[c + s]; __syncthreads(); }
    float var = red[0] * (1.f / Cn);
    out[(size_t)row * Cn + c] = xc * rsqrtf(var + 1e-5f) * w[c] + b[c];
}

__global__ void shift_mix1_kernel(const float* __restrict__ xn, const float* __restrict__ mx,
                                  float* __restrict__ xx, _Float16* __restrict__ xxx)
{
    int row = blockIdx.x, c = threadIdx.x;
    int t = row % Tn;
    size_t rc = (size_t)row * Cn + c;
    float cur  = xn[rc];
    float prev = (t > 0) ? xn[rc - Cn] : 0.f;
    float d = prev - cur;
    xx[rc]  = d;
    xxx[rc] = (_Float16)(cur + d * mx[c]);
}

__global__ void mix_apply5_kernel(const float* __restrict__ xn, const float* __restrict__ xx,
                                  const float* __restrict__ mw, const float* __restrict__ mk,
                                  const float* __restrict__ mv, const float* __restrict__ mr,
                                  const float* __restrict__ mg,
                                  const float* __restrict__ m5, _Float16* __restrict__ xfive)
{
    int row = blockIdx.x, c = threadIdx.x;
    size_t rc = (size_t)row * Cn + c;
    float a = xn[rc], d = xx[rc];
    const float* mm[5] = { mw, mk, mv, mr, mg };
    #pragma unroll
    for (int j = 0; j < 5; ++j) {
        size_t off = (size_t)j * BT * Cn + rc;
        xfive[off] = (_Float16)(a + d * (mm[j][c] + m5[off]));
    }
}

__global__ void decay_transform_kernel(float* __restrict__ wdec, const float* __restrict__ decay)
{
    int row = blockIdx.x, c = threadIdx.x;
    size_t rc = (size_t)row * Cn + c;
    wdec[rc] = __expf(-__expf(decay[c] + wdec[rc]));
}

__global__ void mask_kv_kernel(const int* __restrict__ idx, float* __restrict__ k,
                               float* __restrict__ v)
{
    int row = blockIdx.x, c = threadIdx.x;
    int tok = idx[row];
    float m = (tok != 0 && tok != 1) ? 1.f : 0.f;
    size_t rc = (size_t)row * Cn + c;
    k[rc] *= m; v[rc] *= m;
}

// grid (B*H, 2): y = direction. 64 threads; lane j owns state column j.
__global__ __launch_bounds__(64)
void wkv6_kernel(const float* __restrict__ r, const float* __restrict__ k,
                 const float* __restrict__ v, const float* __restrict__ wd,
                 const float* __restrict__ u, const int* __restrict__ lens,
                 float* __restrict__ yf, float* __restrict__ yb)
{
    int bh = blockIdx.x;
    int dir = blockIdx.y;
    int b = bh / Hn, h = bh % Hn;
    int j = threadIdx.x;
    float* yout = dir ? yb : yf;

    __shared__ float rs[HSn], ks[HSn], ws[HSn], us[HSn];
    us[j] = u[h * HSn + j];
    int len = lens[b];

    float s[HSn];
    #pragma unroll
    for (int i = 0; i < HSn; ++i) s[i] = 0.f;
    __syncthreads();

    for (int tt = 0; tt < Tn; ++tt) {
        int t = tt;
        if (dir) t = (tt < len) ? (len - 1 - tt) : tt;   // involution: read & write pos
        size_t base = ((size_t)(b * Tn + t)) * Cn + h * HSn;
        rs[j] = r[base + j]; ks[j] = k[base + j]; ws[j] = wd[base + j];
        float vj = v[base + j];
        __syncthreads();
        float y = 0.f;
        #pragma unroll
        for (int i = 0; i < HSn; ++i) {
            float kv = ks[i] * vj;
            y += rs[i] * fmaf(us[i], kv, s[i]);
            s[i] = fmaf(ws[i], s[i], kv);
        }
        yout[base + j] = y;
        __syncthreads();
    }
}

// z = groupnorm((yf+yb)*mask) * g  (g already SiLU'd) -> f16 (GEMM A input)
__global__ void gn_gate_kernel(const float* __restrict__ yf, const float* __restrict__ yb,
                               const float* __restrict__ g, const float* __restrict__ gnw,
                               const float* __restrict__ gnb, const int* __restrict__ idx,
                               _Float16* __restrict__ z)
{
    __shared__ float red[Cn];
    int row = blockIdx.x, c = threadIdx.x;
    size_t rc = (size_t)row * Cn + c;
    int tok = idx[row];
    float m = (tok != 0 && tok != 1) ? 1.f : 0.f;
    float val = (yf[rc] + yb[rc]) * m;
    int lane = c & (HSn - 1);
    red[c] = val; __syncthreads();
    for (int s = HSn / 2; s > 0; s >>= 1) { if (lane < s) red[c] += red[c + s]; __syncthreads(); }
    float mu = red[c - lane] * (1.f / HSn);
    __syncthreads();
    float xc = val - mu;
    red[c] = xc * xc; __syncthreads();
    for (int s = HSn / 2; s > 0; s >>= 1) { if (lane < s) red[c] += red[c + s]; __syncthreads(); }
    float var = red[c - lane] * (1.f / HSn);
    float yn = xc * rsqrtf(var + GN_EPS) * gnw[c] + gnb[c];
    z[rc] = (_Float16)(yn * g[rc]);
}

__global__ void ffn_shift_kernel(const float* __restrict__ xn, const float* __restrict__ mk,
                                 const float* __restrict__ mr, _Float16* __restrict__ xkf,
                                 _Float16* __restrict__ xrf)
{
    int row = blockIdx.x, c = threadIdx.x;
    int t = row % Tn;
    size_t rc = (size_t)row * Cn + c;
    float cur  = xn[rc];
    float prev = (t > 0) ? xn[rc - Cn] : 0.f;
    float d = prev - cur;
    xkf[rc] = (_Float16)(cur + d * mk[c]);
    xrf[rc] = (_Float16)(cur + d * mr[c]);
}

__global__ void ffn_add_kernel(float* __restrict__ x, const float* __restrict__ rr,
                               const float* __restrict__ kv)
{
    int row = blockIdx.x, c = threadIdx.x;
    size_t rc = (size_t)row * Cn + c;
    x[rc] += rr[rc] * kv[rc];
}

// ---------------- host side ----------------

template <int ACT, bool ACCUM, bool HOUT, int TMW, int TNW>
static void gemmL(const _Float16* A, int lda, const _Float16* Bt, int ldb,
                  void* C, int ldc, int N, int K, hipStream_t s)
{
    dim3 grid(N / (32 * TNW), BT / (64 * TMW)), blk(256);
    wmma_gemm<ACT, ACCUM, HOUT, TMW, TNW><<<grid, blk, 0, s>>>(A, lda, Bt, ldb, C, ldc, K);
}

static void transposeW(const float* in, _Float16* out, int K, int N, hipStream_t s)
{
    dim3 grid(N / 32, K / 32), blk(256);
    transpose_f16_kernel<<<grid, blk, 0, s>>>(in, out, K, N);
}

extern "C" void kernel_launch(void* const* d_in, const int* in_sizes, int n_in,
                              void* d_out, int out_size, void* d_ws, size_t ws_size,
                              hipStream_t stream)
{
    // setup_inputs() insertion order (recursively flattened):
    // 0 idx | 1 emb | 2 ln0_w | 3 ln0_b | 4 lnout_w | 5 lnout_b
    // layers: 6 ln1_w 7 ln1_b 8 ln2_w 9 ln2_b 10 maa_x 11 maa_w 12 maa_k 13 maa_v
    // 14 maa_r 15 maa_g 16 maa_w1 17 maa_w2 18 decay 19 dec_w1 20 dec_w2 21 faaaa
    // 22 Wr 23 Wk 24 Wv 25 Wg 26 Wo 27 gn_w 28 gn_b 29 fmaa_k 30 fmaa_r
    // 31 Wfk 32 Wfr 33 Wfv
    const int*   idx     = (const int*)  d_in[0];
    const float* emb     = (const float*)d_in[1];
    const float* ln0_w   = (const float*)d_in[2];
    const float* ln0_b   = (const float*)d_in[3];
    const float* lnout_w = (const float*)d_in[4];
    const float* lnout_b = (const float*)d_in[5];

    // workspace layout (bytes, 256-aligned chunks)
    char* wsb = (char*)d_ws;
    size_t off = 0;
    auto alloc = [&](size_t bytes) {
        void* p = wsb + off;
        off = (off + bytes + 255) & ~(size_t)255;
        return p;
    };
    const size_t BTC = (size_t)BT * Cn;
    // fp32 activations
    float* x    = (float*)alloc(BTC * 4);
    float* xn   = (float*)alloc(BTC * 4);
    float* xx   = (float*)alloc(BTC * 4);
    float* m5   = (float*)alloc(5 * BTC * 4);
    float* rb   = (float*)alloc(BTC * 4);
    float* kb   = (float*)alloc(BTC * 4);
    float* vb   = (float*)alloc(BTC * 4);
    float* gb   = (float*)alloc(BTC * 4);
    float* wdec = (float*)alloc(BTC * 4);
    float* yf   = (float*)alloc(BTC * 4);
    float* yb   = (float*)alloc(BTC * 4);
    float* kvb  = (float*)alloc(BTC * 4);
    float* rrb  = (float*)alloc(BTC * 4);
    // f16 activations (GEMM A inputs)
    _Float16* xxx   = (_Float16*)alloc(BTC * 2);
    _Float16* xfive = (_Float16*)alloc(5 * BTC * 2);
    _Float16* hh    = (_Float16*)alloc((size_t)BT * 5 * TMIX * 2);
    _Float16* hdec  = (_Float16*)alloc((size_t)BT * TDEC * 2);
    _Float16* zb    = (_Float16*)alloc(BTC * 2);
    _Float16* kk    = (_Float16*)alloc((size_t)BT * Fn * 2);
    _Float16* xkf   = (_Float16*)alloc(BTC * 2);
    _Float16* xrf   = (_Float16*)alloc(BTC * 2);
    // f16 transposed weights (reused each layer)
    _Float16* w1T  = (_Float16*)alloc((size_t)160 * Cn * 2);
    _Float16* w2T  = (_Float16*)alloc((size_t)Cn * 160 * 2);
    _Float16* dw1T = (_Float16*)alloc((size_t)TDEC * Cn * 2);
    _Float16* dw2T = (_Float16*)alloc((size_t)Cn * TDEC * 2);
    _Float16* WrT  = (_Float16*)alloc((size_t)Cn * Cn * 2);
    _Float16* WkT  = (_Float16*)alloc((size_t)Cn * Cn * 2);
    _Float16* WvT  = (_Float16*)alloc((size_t)Cn * Cn * 2);
    _Float16* WgT  = (_Float16*)alloc((size_t)Cn * Cn * 2);
    _Float16* WoT  = (_Float16*)alloc((size_t)Cn * Cn * 2);
    _Float16* WfkT = (_Float16*)alloc((size_t)Fn * Cn * 2);
    _Float16* WfrT = (_Float16*)alloc((size_t)Cn * Cn * 2);
    _Float16* WfvT = (_Float16*)alloc((size_t)Cn * Fn * 2);
    int* lens = (int*)alloc(Bn * sizeof(int));

    dim3 rows(BT), colsC(Cn);

    lens_kernel<<<Bn, Tn, 0, stream>>>(idx, lens);
    embed_kernel<<<rows, colsC, 0, stream>>>(idx, emb, x);
    layernorm_kernel<<<rows, colsC, 0, stream>>>(x, x, ln0_w, ln0_b);

    for (int l = 0; l < Ln; ++l) {
        const float* ln1_w  = (const float*)d_in[6]  + (size_t)l * Cn;
        const float* ln1_b  = (const float*)d_in[7]  + (size_t)l * Cn;
        const float* ln2_w  = (const float*)d_in[8]  + (size_t)l * Cn;
        const float* ln2_b  = (const float*)d_in[9]  + (size_t)l * Cn;
        const float* maa_x  = (const float*)d_in[10] + (size_t)l * Cn;
        const float* maa_w  = (const float*)d_in[11] + (size_t)l * Cn;
        const float* maa_k  = (const float*)d_in[12] + (size_t)l * Cn;
        const float* maa_v  = (const float*)d_in[13] + (size_t)l * Cn;
        const float* maa_r  = (const float*)d_in[14] + (size_t)l * Cn;
        const float* maa_g  = (const float*)d_in[15] + (size_t)l * Cn;
        const float* maa_w1 = (const float*)d_in[16] + (size_t)l * Cn * (5 * TMIX);
        const float* maa_w2 = (const float*)d_in[17] + (size_t)l * 5 * TMIX * Cn;
        const float* decay  = (const float*)d_in[18] + (size_t)l * Cn;
        const float* dec_w1 = (const float*)d_in[19] + (size_t)l * Cn * TDEC;
        const float* dec_w2 = (const float*)d_in[20] + (size_t)l * TDEC * Cn;
        const float* faaaa  = (const float*)d_in[21] + (size_t)l * Cn;
        const float* Wr     = (const float*)d_in[22] + (size_t)l * Cn * Cn;
        const float* Wk     = (const float*)d_in[23] + (size_t)l * Cn * Cn;
        const float* Wv     = (const float*)d_in[24] + (size_t)l * Cn * Cn;
        const float* Wg     = (const float*)d_in[25] + (size_t)l * Cn * Cn;
        const float* Wo     = (const float*)d_in[26] + (size_t)l * Cn * Cn;
        const float* gn_w   = (const float*)d_in[27] + (size_t)l * Cn;
        const float* gn_b   = (const float*)d_in[28] + (size_t)l * Cn;
        const float* fmaa_k = (const float*)d_in[29] + (size_t)l * Cn;
        const float* fmaa_r = (const float*)d_in[30] + (size_t)l * Cn;
        const float* Wfk    = (const float*)d_in[31] + (size_t)l * Cn * Fn;
        const float* Wfr    = (const float*)d_in[32] + (size_t)l * Cn * Cn;
        const float* Wfv    = (const float*)d_in[33] + (size_t)l * Fn * Cn;

        // ---- per-layer weight convert+transpose to f16 [N][K] ----
        transposeW(maa_w1, w1T, Cn, 160, stream);       // -> [160][512]
        transposeW(maa_w2, w2T, 160, Cn, stream);       // (5*32 x C) -> [512][160]
        transposeW(dec_w1, dw1T, Cn, TDEC, stream);     // -> [64][512]
        transposeW(dec_w2, dw2T, TDEC, Cn, stream);     // -> [512][64]
        transposeW(Wr, WrT, Cn, Cn, stream);
        transposeW(Wk, WkT, Cn, Cn, stream);
        transposeW(Wv, WvT, Cn, Cn, stream);
        transposeW(Wg, WgT, Cn, Cn, stream);
        transposeW(Wo, WoT, Cn, Cn, stream);
        transposeW(Wfk, WfkT, Cn, Fn, stream);          // -> [2048][512]
        transposeW(Wfr, WfrT, Cn, Cn, stream);
        transposeW(Wfv, WfvT, Fn, Cn, stream);          // -> [512][2048]

        // ---- time-mix ----
        layernorm_kernel<<<rows, colsC, 0, stream>>>(x, xn, ln1_w, ln1_b);
        shift_mix1_kernel<<<rows, colsC, 0, stream>>>(xn, maa_x, xx, xxx);
        // hh = tanh(xxx @ maa_w1)   [BT x 160] f16 out; N=160 -> 64x32 tile variant
        gemmL<ACT_TANH, false, true, 1, 1>(xxx, Cn, w1T, Cn, hh, 160, 160, Cn, stream);
        // m[f] = hh[:, f] @ maa_w2[f]  [BT x C] f32, K=32
        for (int f = 0; f < 5; ++f)
            gemmL<ACT_NONE, false, false, 2, 2>(hh + f * TMIX, 160, w2T + f * TMIX, 160,
                                                m5 + (size_t)f * BTC, Cn, Cn, TMIX, stream);
        mix_apply5_kernel<<<rows, colsC, 0, stream>>>(xn, xx, maa_w, maa_k, maa_v, maa_r,
                                                      maa_g, m5, xfive);
        // projections (xfive slices: 0=w 1=k 2=v 3=r 4=g)
        gemmL<ACT_NONE, false, false, 2, 2>(xfive + 3 * BTC, Cn, WrT, Cn, rb, Cn, Cn, Cn, stream);
        gemmL<ACT_NONE, false, false, 2, 2>(xfive + 1 * BTC, Cn, WkT, Cn, kb, Cn, Cn, Cn, stream);
        gemmL<ACT_NONE, false, false, 2, 2>(xfive + 2 * BTC, Cn, WvT, Cn, vb, Cn, Cn, Cn, stream);
        gemmL<ACT_SILU, false, false, 2, 2>(xfive + 4 * BTC, Cn, WgT, Cn, gb, Cn, Cn, Cn, stream);
        // decay: wdec = exp(-exp(decay + tanh(xw@dec_w1)@dec_w2))
        gemmL<ACT_TANH, false, true, 2, 2>(xfive + 0 * BTC, Cn, dw1T, Cn, hdec, TDEC, TDEC, Cn, stream);
        gemmL<ACT_NONE, false, false, 2, 2>(hdec, TDEC, dw2T, TDEC, wdec, Cn, Cn, TDEC, stream);
        decay_transform_kernel<<<rows, colsC, 0, stream>>>(wdec, decay);
        mask_kv_kernel<<<rows, colsC, 0, stream>>>(idx, kb, vb);
        // bidirectional WKV scan (grid.y = direction)
        wkv6_kernel<<<dim3(Bn * Hn, 2), HSn, 0, stream>>>(rb, kb, vb, wdec, faaaa, lens, yf, yb);
        gn_gate_kernel<<<rows, colsC, 0, stream>>>(yf, yb, gb, gn_w, gn_b, idx, zb);
        // x += z @ Wo
        gemmL<ACT_NONE, true, false, 2, 2>(zb, Cn, WoT, Cn, x, Cn, Cn, Cn, stream);

        // ---- channel-mix FFN ----
        layernorm_kernel<<<rows, colsC, 0, stream>>>(x, xn, ln2_w, ln2_b);
        ffn_shift_kernel<<<rows, colsC, 0, stream>>>(xn, fmaa_k, fmaa_r, xkf, xrf);
        // kk = relu(xk @ Wfk)^2  [BT x F] f16 out
        gemmL<ACT_RELUSQ, false, true, 2, 2>(xkf, Cn, WfkT, Cn, kk, Fn, Fn, Cn, stream);
        // kv = kk @ Wfv  [BT x C] f32
        gemmL<ACT_NONE, false, false, 2, 2>(kk, Fn, WfvT, Fn, kvb, Cn, Cn, Fn, stream);
        // rr = sigmoid(xr @ Wfr)
        gemmL<ACT_SIGMOID, false, false, 2, 2>(xrf, Cn, WfrT, Cn, rrb, Cn, Cn, Cn, stream);
        ffn_add_kernel<<<rows, colsC, 0, stream>>>(x, rrb, kvb);
    }

    layernorm_kernel<<<rows, colsC, 0, stream>>>(x, (float*)d_out, lnout_w, lnout_b);
}
